// Transformer_D_55542517072408
// MI455X (gfx1250) — matrix operand order repaired and checked by reading, not compiled, not run on hardware
//
#include <hip/hip_runtime.h>
#include <hip/hip_bf16.h>

typedef __attribute__((ext_vector_type(16))) _Float16 v16h;
typedef __attribute__((ext_vector_type(8)))  _Float16 v8h;
typedef __attribute__((ext_vector_type(8)))  float    v8f;

#define NB   512   // S / C
#define NH   8
#define NDH  32
#define NS   2048
#define NBATCH 4
#define NDIM 256

// ---------------- pack f32 weight [K,N] into WMMA B-fragment layout (f16) ----------
// out[((kt*tilesN + nt)*32 + lane)*16 + e] = W[k,n]
//   k = kt*32 + 8*(lane>>4) + (e&7) + 16*(e>>3) ;  n = nt*16 + (lane&15)
// so each lane's 16 B-halves for one (kt,nt) tile are contiguous (one 32B load).
__global__ void pack_b_kernel(const float* __restrict__ W, _Float16* __restrict__ out,
                              int K, int N) {
    int idx = blockIdx.x * blockDim.x + threadIdx.x;
    if (idx >= K * N) return;
    int e    = idx & 15;
    int L    = (idx >> 4) & 31;
    int rest = idx >> 9;                 // kt*tilesN + nt
    int tilesN = N >> 4;
    int nt = rest % tilesN;
    int kt = rest / tilesN;
    int k  = kt * 32 + ((L >> 4) << 3) + (e & 7) + ((e >> 3) << 4);
    int n  = nt * 16 + (L & 15);
    out[idx] = (_Float16)W[(size_t)k * N + n];
}

// ---------------- LayerNorm: one wave per row of 256, f16 output ----------------
__global__ void layernorm_f16_kernel(const float* __restrict__ x, const float* __restrict__ g,
                                     const float* __restrict__ bta, _Float16* __restrict__ out) {
    int row  = blockIdx.x * (blockDim.x >> 5) + (threadIdx.x >> 5);
    int lane = threadIdx.x & 31;
    const float* xr = x + (size_t)row * NDIM;
    float vals[8];
    float s = 0.f, s2 = 0.f;
#pragma unroll
    for (int i = 0; i < 8; ++i) {
        float v = xr[lane + 32 * i];
        vals[i] = v; s += v; s2 += v * v;
    }
#pragma unroll
    for (int off = 16; off; off >>= 1) {
        s  += __shfl_down(s,  off, 32);
        s2 += __shfl_down(s2, off, 32);
    }
    s  = __shfl(s, 0, 32);
    s2 = __shfl(s2, 0, 32);
    float mean = s * (1.f / NDIM);
    float var  = s2 * (1.f / NDIM) - mean * mean;
    float r    = rsqrtf(var + 1e-5f);
#pragma unroll
    for (int i = 0; i < 8; ++i) {
        int c = lane + 32 * i;
        out[(size_t)row * NDIM + c] = (_Float16)((vals[i] - mean) * r * g[c] + bta[c]);
    }
}

// ---------------- WMMA GEMM: C[M,N] = A[M,K] @ Bpacked  (32x32 macro-tile per wave) ----
// A row-major f16; B pre-packed in fragment layout (see pack_b_kernel).
// flags: 1=+bias  2=LeakyReLU(0.01)  4=+resid(f32)  8=write f16 (else f32)
__global__ void gemm_wmma_kernel(const _Float16* __restrict__ A, const _Float16* __restrict__ Bp,
                                 const float* __restrict__ bias, const float* __restrict__ resid,
                                 float* __restrict__ Cf, _Float16* __restrict__ Ch,
                                 int M, int N, int K, int flags) {
    int wave    = blockIdx.x * (blockDim.x >> 5) + (threadIdx.x >> 5);
    int tilesN2 = N >> 5;                              // 32-wide column tiles
    if (wave >= (M >> 5) * tilesN2) return;            // wave-uniform branch
    int tm   = (wave / tilesN2) << 5;
    int tn2  = (wave % tilesN2);
    int lane = threadIdx.x & 31;
    int hlf  = lane >> 4;
    int lr   = lane & 15;
    int tilesN = N >> 4;

    v8f acc00 = {}, acc01 = {}, acc10 = {}, acc11 = {};

    const _Float16* Ar0 = A + (size_t)(tm + lr) * K + hlf * 8;
    const _Float16* Ar1 = Ar0 + (size_t)16 * K;
    const _Float16* Bb  = Bp + ((size_t)(tn2 * 2) * 32 + lane) * 16;

    for (int k0 = 0; k0 < K; k0 += 32) {
        int kt = k0 >> 5;
        if (k0 + 32 < K) {
            __builtin_prefetch(Ar0 + k0 + 32, 0, 3);
            __builtin_prefetch(Bb + (size_t)(kt + 1) * tilesN * 512, 0, 3);
        }
        // A fragments: lane's halves are contiguous: [k0+8h .. +7] and [k0+8h+16 .. +23]
        v8h a00 = *(const v8h*)(Ar0 + k0);
        v8h a01 = *(const v8h*)(Ar0 + k0 + 16);
        v8h a10 = *(const v8h*)(Ar1 + k0);
        v8h a11 = *(const v8h*)(Ar1 + k0 + 16);
        v16h af0 = __builtin_shufflevector(a00, a01, 0,1,2,3,4,5,6,7,8,9,10,11,12,13,14,15);
        v16h af1 = __builtin_shufflevector(a10, a11, 0,1,2,3,4,5,6,7,8,9,10,11,12,13,14,15);
        // B fragments: contiguous 32B per lane from the packed buffer
        const _Float16* Bk = Bb + (size_t)kt * tilesN * 512;
        v16h bf0 = *(const v16h*)(Bk);
        v16h bf1 = *(const v16h*)(Bk + 512);

        acc00 = __builtin_amdgcn_wmma_f32_16x16x32_f16(false, af0, false, bf0, (short)0, acc00, false, false);
        acc01 = __builtin_amdgcn_wmma_f32_16x16x32_f16(false, af0, false, bf1, (short)0, acc01, false, false);
        acc10 = __builtin_amdgcn_wmma_f32_16x16x32_f16(false, af1, false, bf0, (short)0, acc10, false, false);
        acc11 = __builtin_amdgcn_wmma_f32_16x16x32_f16(false, af1, false, bf1, (short)0, acc11, false, false);
    }

#define STORE_TILE(ACC, TI, TJ)                                             \
    {                                                                       \
        _Pragma("unroll")                                                   \
        for (int r = 0; r < 8; ++r) {                                       \
            int m = tm + (TI)*16 + r + hlf * 8;                             \
            int n = tn2 * 32 + (TJ)*16 + lr;                                \
            float v = (ACC)[r];                                             \
            if (flags & 1) v += bias[n];                                    \
            if (flags & 2) v = v > 0.f ? v : 0.01f * v;                     \
            if (flags & 4) v += resid[(size_t)m * N + n];                   \
            if (flags & 8) Ch[(size_t)m * N + n] = (_Float16)v;             \
            else           Cf[(size_t)m * N + n] = v;                       \
        }                                                                   \
    }
    STORE_TILE(acc00, 0, 0)
    STORE_TILE(acc01, 0, 1)
    STORE_TILE(acc10, 1, 0)
    STORE_TILE(acc11, 1, 1)
#undef STORE_TILE
}

// ---------------- gate logits (N=24, not WMMA-shaped) + sigmoid ----------------
__global__ void gates_kernel(const _Float16* __restrict__ h, const float* __restrict__ Wg,
                             float* __restrict__ gout) {
    int idx = blockIdx.x * blockDim.x + threadIdx.x;   // B*S*24
    if (idx >= NBATCH * NS * 24) return;
    int row = idx / 24, c = idx % 24;
    const _Float16* hr = h + (size_t)row * NDIM;
    float acc = 0.f;
    for (int d = 0; d < NDIM; ++d) acc += (float)hr[d] * Wg[(size_t)d * 24 + c];
    gout[idx] = 1.f / (1.f + expf(-acc));
}

// ---------------- mean-pool K/V blocks: kc,vc [B,H,NB,DH] ----------------
__global__ void kcvc_kernel(const _Float16* __restrict__ k16, const _Float16* __restrict__ v16,
                            float* __restrict__ kc, float* __restrict__ vc) {
    int idx = blockIdx.x * blockDim.x + threadIdx.x;   // B*H*NB*DH = 524288
    int d  = idx & 31;
    int n  = (idx >> 5) & (NB - 1);
    int hh = (idx >> 14) & 7;
    int b  = idx >> 17;
    float sk = 0.f, sv = 0.f;
#pragma unroll
    for (int c = 0; c < 4; ++c) {
        size_t off = ((size_t)(b * NS + n * 4 + c) * NDIM) + hh * NDH + d;
        sk += (float)k16[off];
        sv += (float)v16[off];
    }
    kc[idx] = sk * 0.25f;
    vc[idx] = sv * 0.25f;
}

// ---------------- NSA attention: one thread per (b,h,s) query ----------------
__global__ void attn_kernel(const _Float16* __restrict__ q16, const _Float16* __restrict__ k16,
                            const _Float16* __restrict__ v16, const float* __restrict__ kc,
                            const float* __restrict__ vc, const float* __restrict__ gates,
                            _Float16* __restrict__ attn_out) {
    int idx = blockIdx.x * blockDim.x + threadIdx.x;   // B*H*S = 65536
    int s  = idx & (NS - 1);
    int hh = (idx >> 11) & 7;
    int b  = idx >> 14;
    const float scale = 0.1767766952966369f;           // DH^-0.5

    float qv[32];
    size_t qoff = ((size_t)(b * NS + s) * NDIM) + hh * NDH;
#pragma unroll
    for (int d = 0; d < 32; ++d) qv[d] = (float)q16[qoff + d] * scale;

    int own  = s >> 2;
    int nvis = (s + 1) >> 2;                           // blocks fully in the past
    const float* kcb = kc + (size_t)(b * NH + hh) * NB * NDH;
    const float* vcb = vc + (size_t)(b * NH + hh) * NB * NDH;

    // ---- compressed branch: online softmax over visible blocks; track top-1 (n != own)
    float m = -1e30f, l = 0.f;
    float acc[32];
#pragma unroll
    for (int d = 0; d < 32; ++d) acc[d] = 0.f;
    float bestv = -1e30f; int besti = -1;
    for (int n = 0; n < nvis; ++n) {
        const float* kr = kcb + n * NDH;
        float sc = 0.f;
#pragma unroll
        for (int d = 0; d < 32; ++d) sc += qv[d] * kr[d];
        if (n != own && sc > bestv) { bestv = sc; besti = n; }
        float mn   = sc > m ? sc : m;
        float corr = __expf(m - mn);
        float e    = __expf(sc - mn);
        l = l * corr + e;
        const float* vr = vcb + n * NDH;
#pragma unroll
        for (int d = 0; d < 32; ++d) acc[d] = acc[d] * corr + e * vr[d];
        m = mn;
    }
    size_t grow = (size_t)(b * NS + s) * 24;
    float g0 = gates[grow + 0 * NH + hh];
    float g1 = gates[grow + 1 * NH + hh];
    float g2 = gates[grow + 2 * NH + hh];

    float out[32];
    float inv = (nvis > 0) ? (1.f / l) : 0.f;
#pragma unroll
    for (int d = 0; d < 32; ++d) out[d] = g0 * acc[d] * inv;

    if (besti < 0) besti = (own == 0) ? 1 : 0;         // tie behavior of top_k on all-equal scores

    // ---- selection branch: own block + top-1 block, fine-grained causal attention
    {
        float sm = -1e30f, sl = 0.f;
#pragma unroll
        for (int d = 0; d < 32; ++d) acc[d] = 0.f;
        for (int kk = 0; kk < 8; ++kk) {
            int blk = (kk < 4) ? own : besti;
            int pos = blk * 4 + (kk & 3);
            if (pos > s) continue;
            size_t off = ((size_t)(b * NS + pos) * NDIM) + hh * NDH;
            float sc = 0.f;
#pragma unroll
            for (int d = 0; d < 32; ++d) sc += qv[d] * (float)k16[off + d];
            float mn   = sc > sm ? sc : sm;
            float corr = __expf(sm - mn);
            float e    = __expf(sc - mn);
            sl = sl * corr + e;
#pragma unroll
            for (int d = 0; d < 32; ++d) acc[d] = acc[d] * corr + e * (float)v16[off + d];
            sm = mn;
        }
        float si = (sl > 0.f) ? (1.f / sl) : 0.f;
#pragma unroll
        for (int d = 0; d < 32; ++d) out[d] += g1 * acc[d] * si;
    }

    // ---- sliding window branch (W=2: positions s-1, s)
    {
        float sm = -1e30f, sl = 0.f;
#pragma unroll
        for (int d = 0; d < 32; ++d) acc[d] = 0.f;
        for (int j = 0; j < 2; ++j) {
            int pos = s - 1 + j;
            if (pos < 0) continue;
            size_t off = ((size_t)(b * NS + pos) * NDIM) + hh * NDH;
            float sc = 0.f;
#pragma unroll
            for (int d = 0; d < 32; ++d) sc += qv[d] * (float)k16[off + d];
            float mn   = sc > sm ? sc : sm;
            float corr = __expf(sm - mn);
            float e    = __expf(sc - mn);
            sl = sl * corr + e;
#pragma unroll
            for (int d = 0; d < 32; ++d) acc[d] = acc[d] * corr + e * (float)v16[off + d];
            sm = mn;
        }
        float si = (sl > 0.f) ? (1.f / sl) : 0.f;
#pragma unroll
        for (int d = 0; d < 32; ++d) out[d] += g2 * acc[d] * si;
    }

    size_t ooff = ((size_t)(b * NS + s) * NDIM) + hh * NDH;
#pragma unroll
    for (int d = 0; d < 32; ++d) attn_out[ooff + d] = (_Float16)out[d];
}

// =======================================================================
extern "C" void kernel_launch(void* const* d_in, const int* in_sizes, int n_in,
                              void* d_out, int out_size, void* d_ws, size_t ws_size,
                              hipStream_t stream) {
    const float* x_in  = (const float*)d_in[0];
    const float* lnag  = (const float*)d_in[1];
    const float* lnab  = (const float*)d_in[2];
    const float* Wq    = (const float*)d_in[3];
    const float* Wk    = (const float*)d_in[4];
    const float* Wv    = (const float*)d_in[5];
    const float* Wg    = (const float*)d_in[6];
    const float* Wo    = (const float*)d_in[7];
    const float* lnfg  = (const float*)d_in[8];
    const float* lnfb  = (const float*)d_in[9];
    const float* W1    = (const float*)d_in[10];
    const float* b1    = (const float*)d_in[11];
    const float* W2    = (const float*)d_in[12];
    const float* b2    = (const float*)d_in[13];

    const int ROWS = NBATCH * NS;                    // 8192
    size_t off = 0;
    auto carve = [&](size_t bytes) -> void* {
        void* p = (char*)d_ws + off;
        off = (off + bytes + 255) & ~(size_t)255;
        return p;
    };
    float*    xbuf  = (float*)   carve((size_t)ROWS * NDIM * 4);      // 8 MB
    _Float16* h16   = (_Float16*)carve((size_t)ROWS * NDIM * 2);
    _Float16* q16   = (_Float16*)carve((size_t)ROWS * NDIM * 2);
    _Float16* k16   = (_Float16*)carve((size_t)ROWS * NDIM * 2);
    _Float16* v16   = (_Float16*)carve((size_t)ROWS * NDIM * 2);
    _Float16* at16  = (_Float16*)carve((size_t)ROWS * NDIM * 2);
    _Float16* f116  = (_Float16*)carve((size_t)ROWS * 512 * 2);       // 8 MB
    float*    gat   = (float*)   carve((size_t)ROWS * 24 * 4);
    float*    kcb   = (float*)   carve((size_t)NBATCH * NH * NB * NDH * 4);
    float*    vcb   = (float*)   carve((size_t)NBATCH * NH * NB * NDH * 4);
    _Float16* wq16  = (_Float16*)carve((size_t)4 * NDIM * NDIM * 2);
    _Float16* wk16  = (_Float16*)carve((size_t)4 * NDIM * NDIM * 2);
    _Float16* wv16  = (_Float16*)carve((size_t)4 * NDIM * NDIM * 2);
    _Float16* wo16  = (_Float16*)carve((size_t)4 * NDIM * NDIM * 2);
    _Float16* w116  = (_Float16*)carve((size_t)2 * NDIM * 512 * 2);
    _Float16* w216  = (_Float16*)carve((size_t)2 * 512 * NDIM * 2);

    // ---- pack all weights into WMMA fragment layout (f16), per layer slice
    {
        int n = NDIM * NDIM, blk = (n + 255) / 256;
        for (int i = 0; i < 4; ++i) {
            pack_b_kernel<<<blk, 256, 0, stream>>>(Wq + (size_t)i * n, wq16 + (size_t)i * n, NDIM, NDIM);
            pack_b_kernel<<<blk, 256, 0, stream>>>(Wk + (size_t)i * n, wk16 + (size_t)i * n, NDIM, NDIM);
            pack_b_kernel<<<blk, 256, 0, stream>>>(Wv + (size_t)i * n, wv16 + (size_t)i * n, NDIM, NDIM);
            pack_b_kernel<<<blk, 256, 0, stream>>>(Wo + (size_t)i * n, wo16 + (size_t)i * n, NDIM, NDIM);
        }
        int nf = NDIM * 512, blkf = (nf + 255) / 256;
        for (int l = 0; l < 2; ++l) {
            pack_b_kernel<<<blkf, 256, 0, stream>>>(W1 + (size_t)l * nf, w116 + (size_t)l * nf, NDIM, 512);
            pack_b_kernel<<<blkf, 256, 0, stream>>>(W2 + (size_t)l * nf, w216 + (size_t)l * nf, 512, NDIM);
        }
    }

    hipMemcpyAsync(xbuf, x_in, (size_t)ROWS * NDIM * 4, hipMemcpyDeviceToDevice, stream);

    const int LN_BLOCKS = ROWS / 8;                         // 8 rows (waves) per block
    const int G256_B    = (ROWS / 32) * (NDIM / 32) / 8;    // 256 blocks, 8 waves each
    const int G512_B    = (ROWS / 32) * (512 / 32) / 8;     // 512 blocks

    for (int l = 0; l < 2; ++l) {
        for (int j = 0; j < 2; ++j) {
            int i = l * 2 + j;
            layernorm_f16_kernel<<<LN_BLOCKS, 256, 0, stream>>>(
                xbuf, lnag + (size_t)i * NDIM, lnab + (size_t)i * NDIM, h16);

            gemm_wmma_kernel<<<G256_B, 256, 0, stream>>>(
                h16, wq16 + (size_t)i * NDIM * NDIM, nullptr, nullptr, nullptr, q16,
                ROWS, NDIM, NDIM, /*f16 out*/8);
            gemm_wmma_kernel<<<G256_B, 256, 0, stream>>>(
                h16, wk16 + (size_t)i * NDIM * NDIM, nullptr, nullptr, nullptr, k16,
                ROWS, NDIM, NDIM, 8);
            gemm_wmma_kernel<<<G256_B, 256, 0, stream>>>(
                h16, wv16 + (size_t)i * NDIM * NDIM, nullptr, nullptr, nullptr, v16,
                ROWS, NDIM, NDIM, 8);

            gates_kernel<<<(ROWS * 24 + 255) / 256, 256, 0, stream>>>(
                h16, Wg + (size_t)i * NDIM * 24, gat);

            kcvc_kernel<<<(NBATCH * NH * NB * NDH) / 256, 256, 0, stream>>>(
                k16, v16, kcb, vcb);

            attn_kernel<<<(NBATCH * NH * NS) / 256, 256, 0, stream>>>(
                q16, k16, v16, kcb, vcb, gat, at16);

            // out projection + residual (f32, in-place on xbuf: each elem read then written by its own thread)
            gemm_wmma_kernel<<<G256_B, 256, 0, stream>>>(
                at16, wo16 + (size_t)i * NDIM * NDIM, nullptr, xbuf, xbuf, nullptr,
                ROWS, NDIM, NDIM, /*resid*/4);
        }
        // ---- FFN
        layernorm_f16_kernel<<<LN_BLOCKS, 256, 0, stream>>>(
            xbuf, lnfg + (size_t)l * NDIM, lnfb + (size_t)l * NDIM, h16);
        gemm_wmma_kernel<<<G512_B, 256, 0, stream>>>(
            h16, w116 + (size_t)l * NDIM * 512, b1 + (size_t)l * 512, nullptr, nullptr, f116,
            ROWS, 512, NDIM, /*bias+leaky+f16*/1 | 2 | 8);
        gemm_wmma_kernel<<<G256_B, 256, 0, stream>>>(
            f116, w216 + (size_t)l * 512 * NDIM, b2 + (size_t)l * NDIM, xbuf, xbuf, nullptr,
            ROWS, NDIM, 512, /*bias+resid*/1 | 4);
    }

    hipMemcpyAsync(d_out, xbuf, (size_t)ROWS * NDIM * 4, hipMemcpyDeviceToDevice, stream);
}
